// UncertaintyGAT_29695403884947
// MI455X (gfx1250) — matrix operand order repaired and checked by reading, not compiled, not run on hardware
//
#include <hip/hip_runtime.h>
#include <hip/hip_bf16.h>
#include <math.h>

typedef __attribute__((ext_vector_type(16))) __bf16 v16bf;
typedef __attribute__((ext_vector_type(8)))  __bf16 v8bf;
typedef __attribute__((ext_vector_type(8)))  float  v8f;

// ---------------- helpers ----------------

__device__ __forceinline__ float leaky02(float v) { return v > 0.f ? v : 0.2f * v; }

// float atomic max via signed/unsigned int ordering trick (works for mixed signs,
// initialize to -inf). Emits global_atomic_{max_i32,min_u32}.
__device__ __forceinline__ void atomicMaxFloat(float* addr, float val) {
  if (val >= 0.f) atomicMax((int*)addr, __float_as_int(val));
  else            atomicMin((unsigned int*)addr, __float_as_uint(val));
}

// ---------------- generic elementwise kernels ----------------

__global__ void k_cvt_bf16(const float* __restrict__ src, __bf16* __restrict__ dst, int n) {
  int i = blockIdx.x * blockDim.x + threadIdx.x;
  if (i < n) dst[i] = (__bf16)src[i];
}

__global__ void k_fill(float* __restrict__ p, float v, int n) {
  int i = blockIdx.x * blockDim.x + threadIdx.x;
  if (i < n) p[i] = v;
}

// ---------------- WMMA GEMM: H[n,64] = Xb[n,64] @ Wb[64,64]^T ----------------
// One wave per 16-row tile; computes all 4 column tiles (64 outputs), K=64 -> 2 WMMAs/tile.
__global__ __launch_bounds__(256)
void k_gemm_wmma_bf16(const __bf16* __restrict__ Xb, const __bf16* __restrict__ Wb,
                      float* __restrict__ H, int nRows) {
  const int lane  = threadIdx.x & 31;
  const int wave  = threadIdx.x >> 5;
  const int tile  = blockIdx.x * 8 + wave;   // 16-row tile index
  const int row0  = tile * 16;
  if (row0 >= nRows) return;                 // wave-uniform exit (EXEC all-1 for WMMA)
  const int rlane = lane & 15;
  const int half  = lane >> 4;

  // A fragments: lane holds row (row0+rlane); K pattern per ISA 16-bit A 16x32 table.
  const __bf16* arow = Xb + (size_t)(row0 + rlane) * 64;
  v16bf a[2];
#pragma unroll
  for (int t = 0; t < 2; ++t) {
    v8bf lo = *(const v8bf*)(arow + t * 32 + half * 8);
    v8bf hi = *(const v8bf*)(arow + t * 32 + 16 + half * 8);
#pragma unroll
    for (int e = 0; e < 8; ++e) { a[t][e] = lo[e]; a[t][8 + e] = hi[e]; }
  }

#pragma unroll
  for (int jt = 0; jt < 4; ++jt) {
    // B fragment: lane holds column n = rlane -> W row (jt*16+rlane); K pattern matches A (table order).
    const __bf16* brow = Wb + (size_t)(jt * 16 + rlane) * 64;
    v8f c = {};
#pragma unroll
    for (int t = 0; t < 2; ++t) {
      v8bf lo = *(const v8bf*)(brow + t * 32 + half * 8);
      v8bf hi = *(const v8bf*)(brow + t * 32 + 16 + half * 8);
      v16bf b;
#pragma unroll
      for (int e = 0; e < 8; ++e) { b[e] = lo[e]; b[8 + e] = hi[e]; }
      c = __builtin_amdgcn_wmma_f32_16x16x32_bf16(false, a[t], false, b,
                                                  (short)0, c, false, false);
    }
    // D: lane n=rlane, rows m = v + 8*half
    float* dcol = H + (size_t)row0 * 64 + jt * 16 + rlane;
#pragma unroll
    for (int v = 0; v < 8; ++v) dcol[(size_t)(v + half * 8) * 64] = c[v];
  }
}

// ---------------- attention coefficients: el/er per (node, head) ----------------
__global__ void k_attn_coef(const float* __restrict__ H, const float* __restrict__ al,
                            const float* __restrict__ ar, float* __restrict__ EL,
                            float* __restrict__ ER, int n) {
  int t = blockIdx.x * blockDim.x + threadIdx.x;   // over n*4
  if (t >= n * 4) return;
  int i = t >> 2, h = t & 3;
  const float* row = H + (size_t)i * 64 + h * 16;
  float el = 0.f, er = 0.f;
#pragma unroll
  for (int f = 0; f < 16; ++f) { el += row[f] * al[h * 16 + f]; er += row[f] * ar[h * 16 + f]; }
  EL[t] = el; ER[t] = er;
}

// ---------------- edge softmax passes (4 heads) ----------------
__global__ void k_edge_max(const int* __restrict__ src, const int* __restrict__ dst,
                           const float* __restrict__ EL, const float* __restrict__ ER,
                           float* __restrict__ EMAX, int E) {
  int t = blockIdx.x * blockDim.x + threadIdx.x;   // over E*4
  if (t >= E * 4) return;
  int e = t >> 2, h = t & 3;
  int s = src[e], d = dst[e];
  float v = leaky02(EL[s * 4 + h] + ER[d * 4 + h]);
  atomicMaxFloat(&EMAX[d * 4 + h], v);
}

__global__ void k_edge_sum(const int* __restrict__ src, const int* __restrict__ dst,
                           const float* __restrict__ EL, const float* __restrict__ ER,
                           const float* __restrict__ EMAX, float* __restrict__ DEN, int E) {
  int t = blockIdx.x * blockDim.x + threadIdx.x;
  if (t >= E * 4) return;
  int e = t >> 2, h = t & 3;
  int s = src[e], d = dst[e];
  float v = leaky02(EL[s * 4 + h] + ER[d * 4 + h]);
  atomicAdd(&DEN[d * 4 + h], __expf(v - EMAX[d * 4 + h]));
}

__global__ void k_edge_aggr(const int* __restrict__ src, const int* __restrict__ dst,
                            const float* __restrict__ EL, const float* __restrict__ ER,
                            const float* __restrict__ EMAX, const float* __restrict__ DEN,
                            const float* __restrict__ H, float* __restrict__ OUT,
                            long long total) {
  long long t = (long long)blockIdx.x * blockDim.x + threadIdx.x;  // over E*64
  if (t >= total) return;
  int e = (int)(t >> 6), c = (int)(t & 63), h = c >> 4;
  int s = src[e], d = dst[e];
  float v = leaky02(EL[s * 4 + h] + ER[d * 4 + h]);
  float coef = __expf(v - EMAX[d * 4 + h]) / DEN[d * 4 + h];
  atomicAdd(&OUT[(size_t)d * 64 + c], coef * H[(size_t)s * 64 + c]);
}

// bias + relu in place, re-encode bf16 for next layer's WMMA
__global__ void k_finalize(float* __restrict__ Hio, const float* __restrict__ b,
                           __bf16* __restrict__ Xb, int total) {
  int t = blockIdx.x * blockDim.x + threadIdx.x;   // over n*64
  if (t >= total) return;
  float v = Hio[t] + b[t & 63];
  v = v > 0.f ? v : 0.f;
  Hio[t] = v;
  Xb[t] = (__bf16)v;
}

// ---------------- output layer (1 head, 1 feature) ----------------
__global__ void k_out_proj(const float* __restrict__ H, const float* __restrict__ Wo,
                           const float* __restrict__ alo, const float* __restrict__ aro,
                           float* __restrict__ HO, float* __restrict__ ELo,
                           float* __restrict__ ERo, int n) {
  int i = blockIdx.x * blockDim.x + threadIdx.x;
  if (i >= n) return;
  const float* row = H + (size_t)i * 64;
  float acc = 0.f;
#pragma unroll
  for (int k = 0; k < 64; ++k) acc += row[k] * Wo[k];
  HO[i] = acc; ELo[i] = acc * alo[0]; ERo[i] = acc * aro[0];
}

__global__ void k_edge_max1(const int* __restrict__ src, const int* __restrict__ dst,
                            const float* __restrict__ EL, const float* __restrict__ ER,
                            float* __restrict__ EMAX, int E) {
  int e = blockIdx.x * blockDim.x + threadIdx.x;
  if (e >= E) return;
  atomicMaxFloat(&EMAX[dst[e]], leaky02(EL[src[e]] + ER[dst[e]]));
}

__global__ void k_edge_sum1(const int* __restrict__ src, const int* __restrict__ dst,
                            const float* __restrict__ EL, const float* __restrict__ ER,
                            const float* __restrict__ EMAX, float* __restrict__ DEN, int E) {
  int e = blockIdx.x * blockDim.x + threadIdx.x;
  if (e >= E) return;
  int d = dst[e];
  atomicAdd(&DEN[d], __expf(leaky02(EL[src[e]] + ER[d]) - EMAX[d]));
}

__global__ void k_edge_aggr1(const int* __restrict__ src, const int* __restrict__ dst,
                             const float* __restrict__ EL, const float* __restrict__ ER,
                             const float* __restrict__ EMAX, const float* __restrict__ DEN,
                             const float* __restrict__ HO, float* __restrict__ OUT, int E) {
  int e = blockIdx.x * blockDim.x + threadIdx.x;
  if (e >= E) return;
  int s = src[e], d = dst[e];
  float coef = __expf(leaky02(EL[s] + ER[d]) - EMAX[d]) / DEN[d];
  atomicAdd(&OUT[d], coef * HO[s]);
}

__global__ void k_bias1(float* __restrict__ out, const float* __restrict__ bo, int n) {
  int i = blockIdx.x * blockDim.x + threadIdx.x;
  if (i < n) out[i] += bo[0];
}

// ---------------- launcher ----------------

static inline int nblk(long long n, int b) { return (int)((n + b - 1) / b); }

extern "C" void kernel_launch(void* const* d_in, const int* in_sizes, int n_in,
                              void* d_out, int out_size, void* d_ws, size_t ws_size,
                              hipStream_t stream) {
  const float* x   = (const float*)d_in[0];
  const int*   src = (const int*)d_in[1];
  const int*   dst = (const int*)d_in[2];
  const float* W0  = (const float*)d_in[3];
  const float* al0 = (const float*)d_in[4];
  const float* ar0 = (const float*)d_in[5];
  const float* b0  = (const float*)d_in[6];
  const float* W1  = (const float*)d_in[7];
  const float* al1 = (const float*)d_in[8];
  const float* ar1 = (const float*)d_in[9];
  const float* b1  = (const float*)d_in[10];
  const float* Wo  = (const float*)d_in[11];
  const float* alo = (const float*)d_in[12];
  const float* aro = (const float*)d_in[13];
  const float* bo  = (const float*)d_in[14];
  float* out = (float*)d_out;

  const int N = in_sizes[0] / 64;
  const int E = in_sizes[1];

  // workspace layout (all 16-byte aligned given N multiple of 4)
  float*  H0   = (float*)d_ws;            // N*64  gemm result / scratch
  float*  H1   = H0 + (size_t)N * 64;     // N*64  aggregation accumulator / activations
  float*  EL   = H1 + (size_t)N * 64;     // N*4
  float*  ER   = EL + (size_t)N * 4;      // N*4
  float*  EMAX = ER + (size_t)N * 4;      // N*4
  float*  DEN  = EMAX + (size_t)N * 4;    // N*4
  __bf16* XB   = (__bf16*)(DEN + (size_t)N * 4);  // N*64 bf16
  __bf16* WB   = XB + (size_t)N * 64;             // 64*64 bf16

  const int B = 256;
  const long long NH = (long long)N * 4, EH = (long long)E * 4;
  const long long NF = (long long)N * 64, EF = (long long)E * 64;
  const int gemmBlocks = nblk((N + 15) / 16, 8);
  const float NEG_INF = -__builtin_inff();

  const float* Ws[2]  = {W0, W1};
  const float* als[2] = {al0, al1};
  const float* ars[2] = {ar0, ar1};
  const float* bs[2]  = {b0, b1};

  // encode layer-0 input
  k_cvt_bf16<<<nblk(NF, B), B, 0, stream>>>(x, XB, (int)NF);

  for (int l = 0; l < 2; ++l) {
    k_cvt_bf16<<<nblk(4096, B), B, 0, stream>>>(Ws[l], WB, 4096);
    k_gemm_wmma_bf16<<<gemmBlocks, B, 0, stream>>>(XB, WB, H0, N);
    k_attn_coef<<<nblk(NH, B), B, 0, stream>>>(H0, als[l], ars[l], EL, ER, N);
    k_fill<<<nblk(NH, B), B, 0, stream>>>(EMAX, NEG_INF, (int)NH);
    k_fill<<<nblk(NH, B), B, 0, stream>>>(DEN, 0.f, (int)NH);
    k_fill<<<nblk(NF, B), B, 0, stream>>>(H1, 0.f, (int)NF);
    k_edge_max<<<nblk(EH, B), B, 0, stream>>>(src, dst, EL, ER, EMAX, E);
    k_edge_sum<<<nblk(EH, B), B, 0, stream>>>(src, dst, EL, ER, EMAX, DEN, E);
    k_edge_aggr<<<nblk(EF, B), B, 0, stream>>>(src, dst, EL, ER, EMAX, DEN, H0, H1, EF);
    // bias + relu, store fp32 acts in H1 and bf16 copy in XB for next GEMM
    k_finalize<<<nblk(NF, B), B, 0, stream>>>(H1, bs[l], XB, (int)NF);
  }

  // output layer: scalar head (reuse H0 region for ho/elo/ero, EMAX/DEN first N)
  float* HOs = H0;
  float* ELo = H0 + N;
  float* ERo = H0 + 2 * (size_t)N;
  k_out_proj<<<nblk(N, B), B, 0, stream>>>(H1, Wo, alo, aro, HOs, ELo, ERo, N);
  k_fill<<<nblk(N, B), B, 0, stream>>>(EMAX, NEG_INF, N);
  k_fill<<<nblk(N, B), B, 0, stream>>>(DEN, 0.f, N);
  k_fill<<<nblk(N, B), B, 0, stream>>>(out, 0.f, N);
  k_edge_max1<<<nblk(E, B), B, 0, stream>>>(src, dst, ELo, ERo, EMAX, E);
  k_edge_sum1<<<nblk(E, B), B, 0, stream>>>(src, dst, ELo, ERo, EMAX, DEN, E);
  k_edge_aggr1<<<nblk(E, B), B, 0, stream>>>(src, dst, ELo, ERo, EMAX, DEN, HOs, out, E);
  k_bias1<<<nblk(N, B), B, 0, stream>>>(out, bo, N);
}
